// BoundaryLoss_4861902979372
// MI455X (gfx1250) — compile-verified, matrix operand order checked by reading, not run on hardware
//
#include <hip/hip_runtime.h>

// Problem constants (from reference)
#define BB 2
#define CC 4
#define DDIM 64
#define HDIM 96
#define WDIM 96
constexpr int S3  = DDIM * HDIM * WDIM;   // 589824 spatial voxels per batch
constexpr int NS  = BB * S3;              // 1179648 dist elements (== 2*S3)
constexpr long NL = (long)BB * CC * S3;   // 4718592 logit elements (== 4*NS)
constexpr float FINF = 1e10f;

// Reduction sweep geometry: 2304 waves x (16 slots * 32 lanes * 4 fragments)
// = 2304 * 2048 = 4718592 == NL exactly -> no bounds checks anywhere.
constexpr int RBLK = 288;                 // 288 blocks * 8 waves = 2304 waves

typedef __attribute__((ext_vector_type(16))) _Float16 v16h;
typedef __attribute__((ext_vector_type(8)))  float    v8f;

// Async global->LDS copy of one DWORD per lane (CDNA5 GLOBAL_LOAD_ASYNC_TO_LDS_B32,
// tracked by ASYNCcnt). ldsOff = LDS byte offset (low 32 bits of generic LDS ptr).
__device__ __forceinline__ void async_load_b32(unsigned ldsOff, const float* gptr) {
    unsigned long long ga = (unsigned long long)(uintptr_t)gptr;
    asm volatile("global_load_async_to_lds_b32 %0, %1, off"
                 :: "v"(ldsOff), "v"(ga) : "memory");
}
__device__ __forceinline__ void async_wait0() {
    asm volatile("s_wait_asynccnt 0x0" ::: "memory");
}

// ---------------------------------------------------------------------------
// 1) f0 = 0 where target is a valid class (one-hot sum == 1), else INF
// ---------------------------------------------------------------------------
__global__ void k_init(const int* __restrict__ tgt, float* __restrict__ f) {
    int i = blockIdx.x * blockDim.x + threadIdx.x;
    if (i < NS) {
        int t = tgt[i];
        f[i] = (t >= 0 && t < CC) ? 0.0f : FINF;
    }
}

// ---------------------------------------------------------------------------
// 2) EDT pass along a strided (non-innermost) axis.
//    Array viewed as [outer, n, inner]; block handles 32 inner positions and
//    the full line of length n (n <= 96) staged into LDS via async copies.
//    d[i] = min_j f[j] + (i-j)^2
// ---------------------------------------------------------------------------
__global__ void k_edt_strided(const float* __restrict__ in, float* __restrict__ out,
                              int n, int inner) {
    __shared__ float lds[96 * 32];            // up to 12 KB
    const int  innerBase = blockIdx.x * 32;
    const long outerIdx  = blockIdx.y;
    const float* base  = in  + outerIdx * (long)n * inner + innerBase;
    float*       obase = out + outerIdx * (long)n * inner + innerBase;

    const int tot = n * 32;
    // Stage all lines into LDS asynchronously (memory -> LDS, no VGPR round-trip)
    for (int t = threadIdx.x; t < tot; t += blockDim.x) {
        int j = t >> 5, ii = t & 31;
        async_load_b32((unsigned)(uintptr_t)&lds[t], base + (long)j * inner + ii);
    }
    async_wait0();
    __syncthreads();
    for (int t = threadIdx.x; t < tot; t += blockDim.x) {
        int i = t >> 5, ii = t & 31;
        float best = 3.0e38f;
        for (int j = 0; j < n; ++j) {
            int d = i - j;
            float v = lds[(j << 5) + ii] + (float)(d * d);
            best = fminf(best, v);
        }
        obase[(long)i * inner + ii] = best;
    }
}

// ---------------------------------------------------------------------------
// 3) EDT pass along the contiguous innermost axis (W). One line per block.
// ---------------------------------------------------------------------------
__global__ void k_edt_contig(const float* __restrict__ in, float* __restrict__ out) {
    __shared__ float lds[WDIM];
    const long line = blockIdx.x;
    const int  i    = threadIdx.x;            // 0..95
    async_load_b32((unsigned)(uintptr_t)&lds[i], in + line * WDIM + i);
    async_wait0();
    __syncthreads();
    float best = 3.0e38f;
    for (int j = 0; j < WDIM; ++j) {
        int d = i - j;
        best = fminf(best, lds[j] + (float)(d * d));
    }
    out[line * WDIM + i] = best;
}

// ---------------------------------------------------------------------------
// 4) WMMA weighted reduction:  sum_e sigmoid(logits[e]) * sqrt(fsq[b,s])
//    via D = A x onesT + C with v_wmma_f32_16x16x32_f16 (hi/lo f16 split of
//    each f32 product for ~f32 accuracy, accumulated in f32).
//    Each lane owns 16 contiguous elements per fragment (4x b128 loads);
//    e = it*NS + q, NS = 2*S3  =>  b = (it>=2) [wave-uniform], s = q mod S3
//    (one compare-subtract per lane, block never straddles S3: S3 % 16 == 0).
//    Grid covers NL exactly: zero bounds checks, zero divergence at WMMA.
// ---------------------------------------------------------------------------
__global__ void k_reduce(const float* __restrict__ logits, const float* __restrict__ fsq,
                         float* __restrict__ partials) {
    __shared__ float wpart[8];
    const int lane = threadIdx.x & 31;
    const int wid  = threadIdx.x >> 5;
    const int gwave = blockIdx.x * 8 + wid;   // 0 .. 2303

    const int q0  = gwave * 512 + lane * 16;  // first of this lane's 16 elements, in [0, NS)
    const int qlo = (q0 >= S3) ? (q0 - S3) : q0;  // q0 mod S3 (block stays on one side)

    v16h ones;
#pragma unroll
    for (int k = 0; k < 16; ++k) ones[k] = (_Float16)1.0f;

    v8f acc = {};
#pragma unroll
    for (int half = 0; half < 2; ++half) {    // half == batch index b
        // 16 dist values shared by both class-planes of this batch
        const float4* fp = (const float4*)(fsq + half * S3 + qlo);
        float dv[16];
#pragma unroll
        for (int v = 0; v < 4; ++v) {
            float4 f4 = fp[v];
            dv[4 * v + 0] = __builtin_amdgcn_sqrtf(f4.x);
            dv[4 * v + 1] = __builtin_amdgcn_sqrtf(f4.y);
            dv[4 * v + 2] = __builtin_amdgcn_sqrtf(f4.z);
            dv[4 * v + 3] = __builtin_amdgcn_sqrtf(f4.w);
        }
#pragma unroll
        for (int sub = 0; sub < 2; ++sub) {   // it = half*2 + sub
            const float4* lp = (const float4*)(logits + (long)(half * 2 + sub) * NS + q0);
            v16h ahi, alo;
#pragma unroll
            for (int v = 0; v < 4; ++v) {
                float4 l4 = lp[v];
                float xs[4];
                xs[0] = l4.x; xs[1] = l4.y; xs[2] = l4.z; xs[3] = l4.w;
#pragma unroll
                for (int c = 0; c < 4; ++c) {
                    // fast sigmoid: v_exp_f32 + v_rcp_f32 (~1 ulp)
                    float p = __builtin_amdgcn_rcpf(1.0f + __expf(-xs[c]));
                    float x = p * dv[4 * v + c];
                    _Float16 hi = (_Float16)x;
                    ahi[4 * v + c] = hi;
                    alo[4 * v + c] = (_Float16)(x - (float)hi);
                }
            }
            acc = __builtin_amdgcn_wmma_f32_16x16x32_f16(false, ahi, false, ones,
                                                         (short)0, acc, false, false);
            acc = __builtin_amdgcn_wmma_f32_16x16x32_f16(false, alo, false, ones,
                                                         (short)0, acc, false, false);
        }
    }

    // Sum of all 256 accumulator entries == 16 * (sum of A terms)
    float s = acc[0] + acc[1] + acc[2] + acc[3] + acc[4] + acc[5] + acc[6] + acc[7];
#pragma unroll
    for (int off = 16; off > 0; off >>= 1) s += __shfl_xor(s, off, 32);
    if (lane == 0) wpart[wid] = s * (1.0f / 16.0f);
    __syncthreads();
    if (threadIdx.x == 0) {
        float t = 0.0f;
        for (int w = 0; w < 8; ++w) t += wpart[w];
        partials[blockIdx.x] = t;
    }
}

// ---------------------------------------------------------------------------
// 5) Final deterministic fold of per-block partials, scale, write scalar.
// ---------------------------------------------------------------------------
__global__ void k_final(const float* __restrict__ partials, int nparts,
                        float* __restrict__ out) {
    __shared__ float wpart[8];
    float s = 0.0f;
    for (int i = threadIdx.x; i < nparts; i += blockDim.x) s += partials[i];
    const int lane = threadIdx.x & 31;
    const int wid  = threadIdx.x >> 5;
#pragma unroll
    for (int off = 16; off > 0; off >>= 1) s += __shfl_xor(s, off, 32);
    if (lane == 0) wpart[wid] = s;
    __syncthreads();
    if (threadIdx.x == 0) {
        float t = 0.0f;
        for (int w = 0; w < (int)(blockDim.x >> 5); ++w) t += wpart[w];
        out[0] = t / (float)((long)NS * CC);   // /dist.size then /nc
    }
}

// ---------------------------------------------------------------------------
extern "C" void kernel_launch(void* const* d_in, const int* in_sizes, int n_in,
                              void* d_out, int out_size, void* d_ws, size_t ws_size,
                              hipStream_t stream) {
    (void)in_sizes; (void)n_in; (void)out_size; (void)ws_size;
    const float* logits  = (const float*)d_in[0];
    const int*   targets = (const int*)d_in[1];
    float* out = (float*)d_out;

    // Workspace layout: f0 | f1 | partials  (~9.4 MB + ~1.2 KB)
    float* f0       = (float*)d_ws;
    float* f1       = f0 + NS;
    float* partials = f1 + NS;

    // binary mask -> squared-EDT seed
    k_init<<<(NS + 255) / 256, 256, 0, stream>>>(targets, f0);

    // Pass axis 0 (B): n=2,  inner=S3,  outer=1
    k_edt_strided<<<dim3(S3 / 32, 1), 256, 0, stream>>>(f0, f1, BB, S3);
    // Pass axis 1 (D): n=64, inner=H*W, outer=B
    k_edt_strided<<<dim3((HDIM * WDIM) / 32, BB), 256, 0, stream>>>(f1, f0, DDIM, HDIM * WDIM);
    // Pass axis 2 (H): n=96, inner=W,   outer=B*D
    k_edt_strided<<<dim3(WDIM / 32, BB * DDIM), 256, 0, stream>>>(f0, f1, HDIM, WDIM);
    // Pass axis 3 (W): contiguous lines
    k_edt_contig<<<BB * DDIM * HDIM, WDIM, 0, stream>>>(f1, f0);

    // sigmoid(logits) . sqrt(f0) via WMMA ones-reduction, then final fold
    k_reduce<<<RBLK, 256, 0, stream>>>(logits, f0, partials);
    k_final<<<1, 256, 0, stream>>>(partials, RBLK, out);
}
